// IECNET_197568496167
// MI455X (gfx1250) — matrix operand order repaired and checked by reading, not compiled, not run on hardware
//
#include <hip/hip_runtime.h>

typedef __attribute__((ext_vector_type(16))) _Float16 v16h;
typedef __attribute__((ext_vector_type(8)))  _Float16 v8h;
typedef __attribute__((ext_vector_type(8)))  float    v8f;
typedef __attribute__((ext_vector_type(4)))  int      v4i;

#define E_NUM   100000
#define ENT_DIM 200
#define OUT_CH  16
#define BATCH   1024
#define FCK     12800      // 2*16*20*20 flattened features
#define KP      224        // K padded to multiple of 32 for logits GEMM
#define NP      208        // fc output padded to multiple of 16
#define EPS_BN  1e-5f

// logits kernel tiling
#define NT_PER_BLK 5                   // 5 N-tiles of 16 cols = 80 cols per block
#define LDS_ROW    240                 // halves per LDS row (224 data + 16 pad, 480B, 32B aligned)

// workspace byte offsets (all 32B aligned)
#define OFF_ENTH  0ul                           // f16 [100000, 224]  = 44,800,000 B
#define OFF_FEATH 44800000ul                    // f16 [1024, 12800]  = 26,214,400 B
#define OFF_FCWH  71014400ul                    // f16 [208, 12800]   =  5,324,800 B
#define OFF_Y     76339200ul                    // f32 [1024, 208]    =    851,968 B
#define OFF_XHH   77191168ul                    // f16 [1024, 224]    =    458,752 B

#define AS1 __attribute__((address_space(1)))
#define AS3 __attribute__((address_space(3)))

#if defined(__has_builtin)
#if __has_builtin(__builtin_amdgcn_global_load_async_to_lds_b128)
#define HAVE_ASYNC_LDS 1
#endif
#endif
#ifndef HAVE_ASYNC_LDS
#define HAVE_ASYNC_LDS 0
#endif

// ---------------------------------------------------------------------------
// K0: ent_emb f32 [100000,200] -> f16 [100000,224] zero-padded in K
__global__ void cvt_ent_kernel(const float* __restrict__ ent, _Float16* __restrict__ out) {
    int idx = blockIdx.x * 256 + threadIdx.x;
    if (idx >= E_NUM * KP) return;
    int r = idx / KP, c = idx - r * KP;
    out[idx] = (c < ENT_DIM) ? (_Float16)ent[(size_t)r * ENT_DIM + c] : (_Float16)0.0f;
}

// K0b: fc_w f32 [200,12800] -> f16 [208,12800] zero-padded rows
__global__ void cvt_fcw_kernel(const float* __restrict__ fcw, _Float16* __restrict__ out) {
    int idx = blockIdx.x * 256 + threadIdx.x;
    if (idx >= NP * FCK) return;
    int r = idx / FCK, c = idx - r * FCK;
    out[idx] = (r < ENT_DIM) ? (_Float16)fcw[(size_t)r * FCK + c] : (_Float16)0.0f;
}

// ---------------------------------------------------------------------------
// K1: per-sample rearrange + bn0 + depthwise 3x3 & 1x9 convs + bn1 + relu -> f16 features
__global__ void conv_feat_kernel(const int* __restrict__ eid, const int* __restrict__ rid,
                                 const float* __restrict__ ent, const float* __restrict__ rel,
                                 const float* __restrict__ fw,  const float* __restrict__ fw1,
                                 const float* __restrict__ b0s, const float* __restrict__ b0b,
                                 const float* __restrict__ b0m, const float* __restrict__ b0v,
                                 const float* __restrict__ b1s, const float* __restrict__ b1b,
                                 const float* __restrict__ b1m, const float* __restrict__ b1v,
                                 _Float16* __restrict__ featH) {
    __shared__ float img[20][20];
    __shared__ float flt[16 * 9];
    __shared__ float flt1[16 * 9];
    __shared__ float a1[16], c1[16];

    const int b = blockIdx.x;
    const int t = threadIdx.x;
    const int e = eid[b];
    const int r = rid[b];

    const float ga0 = b0s[0] * rsqrtf(b0v[0] + EPS_BN);
    const float gc0 = b0b[0] - b0m[0] * ga0;
    if (t < 16) {
        float a = b1s[t] * rsqrtf(b1v[t] + EPS_BN);
        a1[t] = a;
        c1[t] = b1b[t] - b1m[t] * a;
    }
    // build 20x20 image: x=[e;r] view [10,40], chunk/transpose/concat per reference
    for (int idx = t; idx < 400; idx += 256) {
        int i = idx / 20, j = idx - (idx / 20) * 20;
        int a, bb;
        if (j < 10) { if (i < 10) { a = i;      bb = j;      }
                      else        { a = i - 10; bb = j + 20; } }
        else        { a = j - 10;  bb = (i < 10) ? (i + 30) : i; }
        int flat = a * 40 + bb;
        float v = (flat < 200) ? ent[(size_t)e * ENT_DIM + flat]
                               : rel[(size_t)r * ENT_DIM + flat - 200];
        img[i][j] = v * ga0 + gc0;
    }
    for (int idx = t; idx < 144; idx += 256) {
        flt[idx]  = fw [(size_t)r * 144 + idx];
        flt1[idx] = fw1[(size_t)r * 144 + idx];
    }
    __syncthreads();

    for (int fi = t; fi < 2 * OUT_CH * 400; fi += 256) {
        float sum = 0.0f;
        int c, i, j;
        if (fi < 6400) {                               // conv1: 3x3, pad 1
            c = fi / 400; int rem = fi - c * 400; i = rem / 20; j = rem - (rem / 20) * 20;
            #pragma unroll
            for (int di = 0; di < 3; ++di) {
                #pragma unroll
                for (int dj = 0; dj < 3; ++dj) {
                    int ii = i + di - 1, jj = j + dj - 1;
                    if (ii >= 0 && ii < 20 && jj >= 0 && jj < 20)
                        sum += img[ii][jj] * flt[c * 9 + di * 3 + dj];
                }
            }
        } else {                                       // conv2: 1x9, pad (0,4)
            int f2 = fi - 6400;
            c = f2 / 400; int rem = f2 - c * 400; i = rem / 20; j = rem - (rem / 20) * 20;
            #pragma unroll
            for (int tt = 0; tt < 9; ++tt) {
                int jj = j + tt - 4;
                if (jj >= 0 && jj < 20)
                    sum += img[i][jj] * flt1[c * 9 + tt];
            }
        }
        float v = fmaxf(sum * a1[c] + c1[c], 0.0f);    // bn1 + relu
        featH[(size_t)b * FCK + fi] = (_Float16)v;
    }
}

// ---------------------------------------------------------------------------
// K2: FC GEMM  Y[1024,208] = feat[1024,12800] x fcw^T  (f16 WMMA, f32 acc)
__global__ void fc_gemm_kernel(const _Float16* __restrict__ A,   // [1024, FCK]
                               const _Float16* __restrict__ Bm,  // [NP, FCK] (B(k,n)=Bm[n][k])
                               float* __restrict__ Y) {          // [1024, NP]
    const int lane = threadIdx.x & 31;
    const int wave = threadIdx.x >> 5;
    const int mBase = (blockIdx.y * 8 + wave) * 16;
    const int nBase = blockIdx.x * 16;

    const _Float16* aRow = A  + (size_t)(mBase + (lane & 15)) * FCK + ((lane < 16) ? 0 : 16);
    const _Float16* bRow = Bm + (size_t)(nBase + (lane & 15)) * FCK + ((lane < 16) ? 0 : 16);

    v8f acc = {};
    for (int kb = 0; kb < FCK; kb += 32) {
        v8h lo = *(const v8h*)(aRow + kb);
        v8h hi = *(const v8h*)(aRow + kb + 8);
        v16h af = __builtin_shufflevector(lo, hi, 0,1,2,3,4,5,6,7,8,9,10,11,12,13,14,15);
        v16h bf = *(const v16h*)(bRow + kb);
        acc = __builtin_amdgcn_wmma_f32_16x16x32_f16(false, af, false, bf,
                                                     (short)0, acc, false, false);
    }
    const int col  = nBase + (lane & 15);
    const int row0 = mBase + ((lane < 16) ? 0 : 8);
    #pragma unroll
    for (int rr = 0; rr < 8; ++rr)
        Y[(size_t)(row0 + rr) * NP + col] = acc[rr];
}

// ---------------------------------------------------------------------------
// K3: fc_b + bn2 + relu -> f16 activations [1024, 224] (zero padded K)
__global__ void bn2_relu_kernel(const float* __restrict__ Y, const float* __restrict__ fcb,
                                const float* __restrict__ s, const float* __restrict__ bb,
                                const float* __restrict__ m, const float* __restrict__ v,
                                _Float16* __restrict__ XhH) {
    int idx = blockIdx.x * 256 + threadIdx.x;
    if (idx >= BATCH * KP) return;
    int b = idx / KP, o = idx - b * KP;
    _Float16 outv = (_Float16)0.0f;
    if (o < ENT_DIM) {
        float t = Y[(size_t)b * NP + o] + fcb[o];
        float a = s[o] * rsqrtf(v[o] + EPS_BN);
        outv = (_Float16)fmaxf((t - m[o]) * a + bb[o], 0.0f);
    }
    XhH[idx] = outv;
}

// ---------------------------------------------------------------------------
// K4: logits GEMM [1024,224]x[224,100000] + out_bias, sigmoid -> d_out f32
// Block = 8 waves = 8 M-tiles (128 rows) x 5 N-tiles (80 cols).
// B panel (80 rows x 448B, contiguous in EntH) staged once into LDS via the
// CDNA5 async-to-LDS path (ASYNCcnt + s_wait_asynccnt), then shared by all
// 8 waves -> 8x less B traffic through WGP$/L2.
__global__ void logits_kernel(const _Float16* __restrict__ Xh,  // [1024, KP]
                              const _Float16* __restrict__ Wt,  // [E_NUM, KP] (B(k,n)=Wt[n][k])
                              const float* __restrict__ obias,
                              float* __restrict__ out) {        // [1024, E_NUM]
    __shared__ _Float16 Bsh[NT_PER_BLK * 16 * LDS_ROW];         // 38,400 B

    const int lane = threadIdx.x & 31;
    const int wave = threadIdx.x >> 5;
    const int mBase = (blockIdx.y * 8 + wave) * 16;
    const int nRow0 = blockIdx.x * (NT_PER_BLK * 16);           // first EntH row of panel

    // A fragments resident in VGPRs (7 k-steps)
    const _Float16* aRow = Xh + (size_t)(mBase + (lane & 15)) * KP + ((lane < 16) ? 0 : 16);
    v16h af[7];
    #pragma unroll
    for (int ks = 0; ks < 7; ++ks) {
        v8h lo = *(const v8h*)(aRow + ks * 32);
        v8h hi = *(const v8h*)(aRow + ks * 32 + 8);
        af[ks] = __builtin_shufflevector(lo, hi, 0,1,2,3,4,5,6,7,8,9,10,11,12,13,14,15);
    }

    // Cooperative copy of B panel: 80 rows x 28 b128-chunks (448B of data/row);
    // LDS rows padded to 480B so every 32B vector read stays aligned.
    {
        char* gsrc  = (char*)(void*)(Wt + (size_t)nRow0 * KP);
        char* lbase = (char*)Bsh;
        for (int i = threadIdx.x; i < (NT_PER_BLK * 16) * (KP * 2 / 16); i += 256) {
            int row = i / 28, chunk = i - row * 28;
            char* g = gsrc  + (size_t)row * (KP * 2) + chunk * 16;
            char* l = lbase + (size_t)row * (LDS_ROW * 2) + chunk * 16;
#if HAVE_ASYNC_LDS
            __builtin_amdgcn_global_load_async_to_lds_b128(
                (AS1 v4i*)g, (AS3 v4i*)l, 0, 0);
#else
            *(v4i*)l = *(const v4i*)g;
#endif
        }
    }
#if HAVE_ASYNC_LDS
#if defined(__has_builtin) && __has_builtin(__builtin_amdgcn_s_wait_asynccnt)
    __builtin_amdgcn_s_wait_asynccnt(0);
#else
    asm volatile("s_wait_asynccnt 0x0" ::: "memory");
#endif
#endif
    __syncthreads();

    // Each wave: 1 M-tile x 5 N-tiles, B fragments from LDS
    #pragma unroll
    for (int nt = 0; nt < NT_PER_BLK; ++nt) {
        const _Float16* bL = &Bsh[(size_t)(nt * 16 + (lane & 15)) * LDS_ROW +
                                  ((lane < 16) ? 0 : 16)];
        v8f acc = {};
        #pragma unroll
        for (int ks = 0; ks < 7; ++ks) {
            v16h bf = *(const v16h*)(bL + ks * 32);
            acc = __builtin_amdgcn_wmma_f32_16x16x32_f16(false, af[ks], false, bf,
                                                         (short)0, acc, false, false);
        }
        const int col  = nRow0 + nt * 16 + (lane & 15);
        const int row0 = mBase + ((lane < 16) ? 0 : 8);
        const float bias = obias[col];
        #pragma unroll
        for (int rr = 0; rr < 8; ++rr) {
            float x = acc[rr] + bias;
            out[(size_t)(row0 + rr) * E_NUM + col] = 1.0f / (1.0f + __expf(-x));
        }
    }
}

// ---------------------------------------------------------------------------
extern "C" void kernel_launch(void* const* d_in, const int* in_sizes, int n_in,
                              void* d_out, int out_size, void* d_ws, size_t ws_size,
                              hipStream_t stream) {
    const int*   eid   = (const int*)  d_in[0];
    const int*   rid   = (const int*)  d_in[1];
    const float* ent   = (const float*)d_in[2];
    const float* rel   = (const float*)d_in[3];
    const float* fw    = (const float*)d_in[4];
    const float* fw1   = (const float*)d_in[5];
    const float* fcw   = (const float*)d_in[6];
    const float* fcb   = (const float*)d_in[7];
    const float* obias = (const float*)d_in[8];
    const float* b0s   = (const float*)d_in[9];
    const float* b0b   = (const float*)d_in[10];
    const float* b0m   = (const float*)d_in[11];
    const float* b0v   = (const float*)d_in[12];
    const float* b1s   = (const float*)d_in[13];
    const float* b1b   = (const float*)d_in[14];
    const float* b1m   = (const float*)d_in[15];
    const float* b1v   = (const float*)d_in[16];
    const float* b2s   = (const float*)d_in[17];
    const float* b2b   = (const float*)d_in[18];
    const float* b2m   = (const float*)d_in[19];
    const float* b2v   = (const float*)d_in[20];
    (void)in_sizes; (void)n_in; (void)out_size; (void)ws_size;

    char* ws = (char*)d_ws;
    _Float16* EntH  = (_Float16*)(ws + OFF_ENTH);
    _Float16* FeatH = (_Float16*)(ws + OFF_FEATH);
    _Float16* FcwH  = (_Float16*)(ws + OFF_FCWH);
    float*    Yws   = (float*)   (ws + OFF_Y);
    _Float16* XhH   = (_Float16*)(ws + OFF_XHH);
    float*    outp  = (float*)d_out;

    cvt_ent_kernel<<<(E_NUM * KP + 255) / 256, 256, 0, stream>>>(ent, EntH);
    cvt_fcw_kernel<<<(NP * FCK + 255) / 256, 256, 0, stream>>>(fcw, FcwH);
    conv_feat_kernel<<<BATCH, 256, 0, stream>>>(eid, rid, ent, rel, fw, fw1,
                                                b0s, b0b, b0m, b0v,
                                                b1s, b1b, b1m, b1v, FeatH);
    fc_gemm_kernel<<<dim3(NP / 16, 8), 256, 0, stream>>>(FeatH, FcwH, Yws);
    bn2_relu_kernel<<<(BATCH * KP + 255) / 256, 256, 0, stream>>>(Yws, fcb,
                                                b2s, b2b, b2m, b2v, XhH);
    logits_kernel<<<dim3(E_NUM / 16 / NT_PER_BLK, 8), 256, 0, stream>>>(XhH, EntH, obias, outp);
}
